// GroupQueryAttention_32375463477538
// MI455X (gfx1250) — matrix operand order repaired and checked by reading, not compiled, not run on hardware
//
#include <hip/hip_runtime.h>
#include <hip/hip_bf16.h>

typedef __bf16 bf16;
typedef __attribute__((ext_vector_type(16))) __bf16 v16bf;
typedef __attribute__((ext_vector_type(8)))  __bf16 v8bf;
typedef __attribute__((ext_vector_type(8)))  float   v8f;

constexpr int Bb = 2, Ss = 2048, Dd = 2048, Hh = 16, Gg = 8, HDd = 128;
constexpr int NQKV = Hh * HDd + 2 * Gg * HDd;   // 4096 fused qkv width
constexpr float EPSf = 1e-6f;
// softmax scale folded into q, in base-2: (1/sqrt(128)) * log2(e).
// After RMSNorm |q.k| <= 128, so base-2 scores are in [-16.4, 16.4]:
// exp2 never over/underflows in f32 -> max-free softmax is exact.
constexpr float QSCALE2 = 0.08838834764831845f * 1.4426950408889634f;

__device__ __forceinline__ v16bf cat8(v8bf lo, v8bf hi) {
  v16bf r;
#pragma unroll
  for (int i = 0; i < 8; i++) { r[i] = lo[i]; r[8 + i] = hi[i]; }
  return r;
}

// ---------------- x convert (coalesced) ----------------
__global__ void k_f32_to_bf16(const float* __restrict__ in, bf16* __restrict__ out, int n) {
  int i = blockIdx.x * blockDim.x + threadIdx.x;
  if (i < n) out[i] = (bf16)in[i];
}

// ---------------- tiled transpose f32 -> bf16: out[n][k] = in[k][n] ----------------
// grid = (K/32, N/32); 256 threads; coalesced loads and stores via padded LDS tile
__global__ void __launch_bounds__(256) k_transpose_f2b(const float* __restrict__ in,
                                                       bf16* __restrict__ out,
                                                       int ldi, int ldo) {
  __shared__ float tile[32][33];
  const int k0 = blockIdx.x * 32, n0 = blockIdx.y * 32;
  const int c = threadIdx.x & 31, r8 = threadIdx.x >> 5;
#pragma unroll
  for (int i = 0; i < 4; i++)
    tile[r8 + i * 8][c] = in[(size_t)(k0 + r8 + i * 8) * ldi + n0 + c];
  __syncthreads();
#pragma unroll
  for (int i = 0; i < 4; i++)
    out[(size_t)(n0 + r8 + i * 8) * ldo + k0 + c] = (bf16)tile[c][r8 + i * 8];
}

// ---------------- V repack (tiled): qkv v-section -> vt[b][g][hd][s] ----------------
// grid = (S/32, HD/32, B*G)
__global__ void __launch_bounds__(256) k_transpose_v(const bf16* __restrict__ qkv,
                                                     bf16* __restrict__ vt) {
  __shared__ bf16 tile[32][33];
  const int z = blockIdx.z;              // b*G + g
  const int b = z >> 3, g = z & 7;
  const int s0 = blockIdx.x * 32, j0 = blockIdx.y * 32;
  const bf16* in  = qkv + (size_t)b * Ss * NQKV + Hh * HDd + Gg * HDd + g * HDd;
  bf16*       out = vt + (size_t)z * HDd * Ss;
  const int c = threadIdx.x & 31, r8 = threadIdx.x >> 5;
#pragma unroll
  for (int i = 0; i < 4; i++)
    tile[r8 + i * 8][c] = in[(size_t)(s0 + r8 + i * 8) * NQKV + j0 + c];
  __syncthreads();
#pragma unroll
  for (int i = 0; i < 4; i++)
    out[(size_t)(j0 + r8 + i * 8) * Ss + s0 + c] = tile[c][r8 + i * 8];
}

// ---------------- generic bf16 WMMA GEMM: C = A(MxK) * Bt(NxK)^T ----------------
// block = 256 threads = 8 waves; block tile 64(M) x 256(N); wave tile 32x64
template <typename OutT>
__global__ void __launch_bounds__(256) k_gemm_bf16(const bf16* __restrict__ A,
                                                   const bf16* __restrict__ Bt,
                                                   OutT* __restrict__ C,
                                                   int M, int N, int K) {
  const int lane = threadIdx.x & 31;
  const int wid  = threadIdx.x >> 5;
  const int l16  = lane & 15;
  const int half = lane >> 4;
  const int m0 = blockIdx.x * 64 + (wid >> 2) * 32;
  const int n0 = blockIdx.y * 256 + (wid & 3) * 64;

  const bf16* a0 = A + (size_t)(m0 + l16) * K;
  const bf16* a1 = a0 + (size_t)16 * K;
  const bf16* bp[4];
#pragma unroll
  for (int t = 0; t < 4; t++) bp[t] = Bt + (size_t)(n0 + t * 16 + l16) * K;

  v8f acc[2][4] = {};

  for (int k = 0; k < K; k += 32) {
    const int kb = k + half * 8;     // A frag: elems 0..7 = K kb..kb+7, 8..15 = kb+16..kb+23
    v16bf af[2];
    af[0] = cat8(*(const v8bf*)(a0 + kb), *(const v8bf*)(a0 + kb + 16));
    af[1] = cat8(*(const v8bf*)(a1 + kb), *(const v8bf*)(a1 + kb + 16));
    const int kh = k + half * 8;     // B frag: elems 0..7 = K kh..kh+7, 8..15 = kh+16..kh+23
#pragma unroll
    for (int t = 0; t < 4; t++) {
      v16bf bfv = cat8(*(const v8bf*)(bp[t] + kh), *(const v8bf*)(bp[t] + kh + 16));
#pragma unroll
      for (int i = 0; i < 2; i++)
        acc[i][t] = __builtin_amdgcn_wmma_f32_16x16x32_bf16(
            false, af[i], false, bfv, (short)0, acc[i][t], false, false);
    }
  }

#pragma unroll
  for (int i = 0; i < 2; i++)
#pragma unroll
    for (int t = 0; t < 4; t++)
#pragma unroll
      for (int v = 0; v < 8; v++) {
        int row = m0 + i * 16 + half * 8 + v;
        int col = n0 + t * 16 + l16;
        C[(size_t)row * N + col] = (OutT)acc[i][t][v];
      }
}

// ---------------- RMSNorm + RoPE on q and k heads ----------------
// one wave per (b, s, head) 128-vector; first H heads are q, next G are k.
// Base-2 softmax scale folded into q here so the attention hot loop is mul-free.
__global__ void __launch_bounds__(256) k_qk_norm_rope(
    const bf16* __restrict__ qkv, const float* __restrict__ cosT,
    const float* __restrict__ sinT, const float* __restrict__ qscale,
    const float* __restrict__ kscale, bf16* __restrict__ qn, bf16* __restrict__ kn) {
  const int lane = threadIdx.x & 31;
  const int wid  = threadIdx.x >> 5;
  const int NH = Hh + Gg;                       // 24
  int w  = blockIdx.x * 8 + wid;
  int hh = w % NH;
  int s  = (w / NH) % Ss;
  int b  = w / (NH * Ss);
  bool isq = hh < Hh;
  int col0 = isq ? hh * HDd : Hh * HDd + (hh - Hh) * HDd;

  const bf16* src = qkv + (size_t)(b * Ss + s) * NQKV + col0 + lane * 4;
  float x[4];
#pragma unroll
  for (int i = 0; i < 4; i++) x[i] = (float)src[i];

  float ss2 = x[0]*x[0] + x[1]*x[1] + x[2]*x[2] + x[3]*x[3];
#pragma unroll
  for (int m = 1; m < 32; m <<= 1) ss2 += __shfl_xor(ss2, m, 32);
  float rinv = rsqrtf(ss2 * (1.0f / 128.0f) + EPSf);

  const float* scv = (isq ? qscale : kscale) + lane * 4;
  const float* cp  = cosT + (size_t)s * HDd + lane * 4;
  const float* sp  = sinT + (size_t)s * HDd + lane * 4;
  bf16* dst = isq
      ? qn + (((size_t)(b * Hh + hh) * Ss + s) * HDd + lane * 4)
      : kn + (((size_t)(b * Gg + (hh - Hh)) * Ss + s) * HDd + lane * 4);

  const float post = isq ? QSCALE2 : 1.0f;
  float y[4];
#pragma unroll
  for (int i = 0; i < 4; i++) y[i] = x[i] * rinv * scv[i];
#pragma unroll
  for (int i = 0; i < 4; i++) {
    float p   = __shfl_xor(y[i], 16, 32);       // partner from other half of vector
    float rot = (lane < 16) ? -p : p;           // rotate-half
    dst[i] = (bf16)((y[i] * cp[i] + rot * sp[i]) * post);
  }
}

// ---------------- causal GQA flash attention (max-free softmax) ----------------
// one 32-key chunk; MASK=true only for the diagonal chunk.
// No cross-lane ops in the hot loop: only per-lane exp2 + partial-sum accumulate.
template <bool MASK>
__device__ __forceinline__ void attn_chunk(int k0, int q0, int l16, int half,
                                           const v16bf (&qf)[4],
                                           const bf16* __restrict__ kbase,
                                           const bf16* __restrict__ vbase,
                                           bf16* __restrict__ pw,
                                           v8f (&o)[8], float (&lsum)[8]) {
  const float NEG_INF = -__builtin_inff();
  v8f sc[2] = {};
#pragma unroll
  for (int sub = 0; sub < 2; sub++) {
    const bf16* kp = kbase + (size_t)(k0 + sub * 16 + l16) * HDd;
#pragma unroll
    for (int c = 0; c < 4; c++) {
      int kh = c * 32 + half * 8;
      v16bf kf = cat8(*(const v8bf*)(kp + kh), *(const v8bf*)(kp + kh + 16));
      sc[sub] = __builtin_amdgcn_wmma_f32_16x16x32_bf16(
          false, qf[c], false, kf, (short)0, sc[sub], false, false);
    }
  }
  // P = exp2(scores); scores pre-scaled to log2 units and bounded (+-16.4)
  float p0[8], p1[8];
#pragma unroll
  for (int v = 0; v < 8; v++) {
    float s0 = sc[0][v];
    float s1 = sc[1][v];
    if (MASK) {
      int qrow = q0 + half * 8 + v;
      if (k0 + l16 > qrow)      s0 = NEG_INF;
      if (k0 + 16 + l16 > qrow) s1 = NEG_INF;
    }
    float e0 = exp2f(s0), e1 = exp2f(s1);
    lsum[v] += e0 + e1;                 // per-lane partial row sum; reduced once at end
    p0[v] = e0; p1[v] = e1;
  }
  // P (C-layout) -> LDS -> A-fragment layout
#pragma unroll
  for (int v = 0; v < 8; v++) {
    int row = half * 8 + v;
    pw[row * 32 + l16]      = (bf16)p0[v];
    pw[row * 32 + 16 + l16] = (bf16)p1[v];
  }
  asm volatile("s_wait_dscnt 0" ::: "memory");
  const bf16* pr = pw + l16 * 32 + half * 8;
  v16bf pa = cat8(*(const v8bf*)pr, *(const v8bf*)(pr + 16));
  // ctx += P(16x32) @ V(32x128): 8 hd tiles
#pragma unroll
  for (int t = 0; t < 8; t++) {
    const bf16* vp = vbase + (size_t)(t * 16 + l16) * Ss + k0 + half * 8;
    v16bf vf = cat8(*(const v8bf*)vp, *(const v8bf*)(vp + 16));
    o[t] = __builtin_amdgcn_wmma_f32_16x16x32_bf16(
        false, pa, false, vf, (short)0, o[t], false, false);
  }
}

// one wave per (b, h, 16-query tile)
__global__ void __launch_bounds__(256) k_attn(const bf16* __restrict__ qn,
                                              const bf16* __restrict__ kn,
                                              const bf16* __restrict__ vt,
                                              bf16* __restrict__ ctx) {
  __shared__ alignas(16) bf16 smem[8][16 * 32];   // per-wave P-transpose staging
  const int lane = threadIdx.x & 31;
  const int wid  = threadIdx.x >> 5;
  const int l16  = lane & 15;
  const int half = lane >> 4;

  const int QT = Ss / 16;                         // 128 q tiles per head
  int gw = blockIdx.x * 8 + wid;
  int qt = gw % QT;
  int h  = (gw / QT) % Hh;
  int b  = gw / (QT * Hh);
  int g  = h >> 1;                                // GS = H/G = 2
  int q0 = qt * 16;

  // Q fragments for 4 hd-chunks of 32 (softmax scale pre-folded)
  const bf16* qbase = qn + ((size_t)(b * Hh + h) * Ss + q0 + l16) * HDd;
  v16bf qf[4];
#pragma unroll
  for (int c = 0; c < 4; c++) {
    int kb = c * 32 + half * 8;
    qf[c] = cat8(*(const v8bf*)(qbase + kb), *(const v8bf*)(qbase + kb + 16));
  }

  v8f o[8] = {};
  float lsum[8];
#pragma unroll
  for (int v = 0; v < 8; v++) lsum[v] = 0.0f;

  const bf16* kbase = kn + (size_t)(b * Gg + g) * Ss * HDd;
  const bf16* vbase = vt + (size_t)(b * Gg + g) * HDd * Ss;
  bf16* pw = &smem[wid][0];

  const int ninner = q0 / 32;                     // chunks strictly below the diagonal
  for (int kc = 0; kc < ninner; kc++)
    attn_chunk<false>(kc * 32, q0, l16, half, qf, kbase, vbase, pw, o, lsum);
  attn_chunk<true>(ninner * 32, q0, l16, half, qf, kbase, vbase, pw, o, lsum);

  // single row-sum reduction (16-lane butterfly per row), then normalize + store
  float rinv[8];
#pragma unroll
  for (int v = 0; v < 8; v++) {
    float rs = lsum[v];
#pragma unroll
    for (int m = 1; m < 16; m <<= 1) rs += __shfl_xor(rs, m, 16);
    rinv[v] = 1.0f / rs;
  }
#pragma unroll
  for (int t = 0; t < 8; t++)
#pragma unroll
    for (int v = 0; v < 8; v++) {
      int row = q0 + half * 8 + v;
      ctx[(size_t)(b * Ss + row) * Dd + h * HDd + t * 16 + l16] =
          (bf16)(o[t][v] * rinv[v]);
    }
}

// ---------------- launcher ----------------
extern "C" void kernel_launch(void* const* d_in, const int* in_sizes, int n_in,
                              void* d_out, int out_size, void* d_ws, size_t ws_size,
                              hipStream_t stream) {
  (void)in_sizes; (void)n_in; (void)out_size; (void)ws_size;
  const float* x    = (const float*)d_in[0];
  // d_in[1] = mask (bool) unused: causal mask computed analytically
  const float* cosT = (const float*)d_in[2];
  const float* sinT = (const float*)d_in[3];
  const float* Wq   = (const float*)d_in[4];
  const float* Wk   = (const float*)d_in[5];
  const float* Wv   = (const float*)d_in[6];
  const float* Wo   = (const float*)d_in[7];
  const float* qs   = (const float*)d_in[8];
  const float* ks   = (const float*)d_in[9];

  char* ws = (char*)d_ws;                    // ~126 MB total, 16B-aligned offsets
  bf16* xb    = (bf16*)(ws + 0);             // 16.8 MB  x bf16 [4096][2048]
  bf16* wqkvt = (bf16*)(ws + 16777216);      // 16.8 MB  fused W^T [4096][2048]
  bf16* wot   = (bf16*)(ws + 33554432);      //  8.4 MB  Wo^T [2048][2048]
  bf16* qkv   = (bf16*)(ws + 41943040);      // 33.6 MB  qkv [4096][4096]
  bf16* qn    = (bf16*)(ws + 75497472);      // 16.8 MB  q normed/roped [b][h][s][hd]
  bf16* kn    = (bf16*)(ws + 92274688);      //  8.4 MB  k normed/roped [b][g][s][hd]
  bf16* vtb   = (bf16*)(ws + 100663296);     //  8.4 MB  v transposed [b][g][hd][s]
  bf16* ctxb  = (bf16*)(ws + 109051904);     // 16.8 MB  ctx [4096][2048]

  const int M = Bb * Ss;                     // 4096

  k_f32_to_bf16<<<(M * Dd) / 256, 256, 0, stream>>>(x, xb, M * Dd);
  // pack fused [Wq|Wk|Wv]^T and Wo^T with coalesced tiled transposes
  k_transpose_f2b<<<dim3(Dd / 32, 2048 / 32), 256, 0, stream>>>(Wq, wqkvt, 2048, Dd);
  k_transpose_f2b<<<dim3(Dd / 32, 1024 / 32), 256, 0, stream>>>(
      Wk, wqkvt + (size_t)2048 * Dd, 1024, Dd);
  k_transpose_f2b<<<dim3(Dd / 32, 1024 / 32), 256, 0, stream>>>(
      Wv, wqkvt + (size_t)3072 * Dd, 1024, Dd);
  k_transpose_f2b<<<dim3(Dd / 32, Dd / 32), 256, 0, stream>>>(Wo, wot, Dd, Dd);

  k_gemm_bf16<bf16><<<dim3(M / 64, NQKV / 256), 256, 0, stream>>>(
      xb, wqkvt, qkv, M, NQKV, Dd);

  k_qk_norm_rope<<<(Bb * Ss * (Hh + Gg)) / 8, 256, 0, stream>>>(
      qkv, cosT, sinT, qs, ks, qn, kn);
  k_transpose_v<<<dim3(Ss / 32, HDd / 32, Bb * Gg), 256, 0, stream>>>(qkv, vtb);

  k_attn<<<(Bb * Hh * (Ss / 16)) / 8, 256, 0, stream>>>(qn, kn, vtb, ctxb);

  k_gemm_bf16<float><<<dim3(M / 64, Dd / 256), 256, 0, stream>>>(
      ctxb, wot, (float*)d_out, M, Dd, Dd);
}
